// CausalSelfAttention_120259084538
// MI455X (gfx1250) — compile-verified, matrix operand order checked by reading, not run on hardware
//
#include <hip/hip_runtime.h>
#include <hip/hip_bf16.h>

typedef _Float16 h2  __attribute__((ext_vector_type(2)));
typedef _Float16 h8  __attribute__((ext_vector_type(8)));
typedef _Float16 h16 __attribute__((ext_vector_type(16)));
typedef float    f8  __attribute__((ext_vector_type(8)));

#define D_MODEL 2048
#define SEQ     2048
#define NH      16
#define HD      128
#define THREE_D 6144

// Load a 16x32 WMMA operand fragment (A-style) from a row-major LDS tile.
// Per ISA: lane holds row (lane&15); k = (i/8)*16 + bank*8 + (i%8), bank = lane>>4.
__device__ __forceinline__ h16 ldsFrag(const _Float16* base, int row, int stride,
                                       int colOff, int bank) {
    const h8* p0 = reinterpret_cast<const h8*>(base + row * stride + colOff + bank * 8);
    const h8* p1 = reinterpret_cast<const h8*>(base + row * stride + colOff + 16 + bank * 8);
    h8 lo = *p0, hi = *p1;
    h16 r;
#pragma unroll
    for (int i = 0; i < 8; ++i) { r[i] = lo[i]; r[i + 8] = hi[i]; }
    return r;
}

__device__ __forceinline__ f8 wmma32(h16 a, h16 b, f8 c) {
    return __builtin_amdgcn_wmma_f32_16x16x32_f16(false, a, false, b, (short)0, c,
                                                  false, false);
}

// CDNA5 async copy: 16 bytes global -> LDS, tracked by ASYNCcnt (no VGPR round trip).
__device__ __forceinline__ void asyncCopy16(void* ldsDst, const void* gsrc) {
    uint32_t l = (uint32_t)(uintptr_t)ldsDst;   // LDS aperture: low 32 bits = LDS addr
    asm volatile("global_load_async_to_lds_b128 %0, %1, off"
                 :: "v"(l), "v"(gsrc) : "memory");
}
__device__ __forceinline__ void asyncWait() {
    asm volatile("s_wait_asynccnt 0x0" ::: "memory");
}

// ---------------------------------------------------------------------------
// Kernel 1: qkv = x @ qkv_w + b, RoPE(q,k), q *= 1/sqrt(HD); write fp16 [B,H,S,HD]
// grid (48 col-tiles of 128, 64 row-tiles of 64), 256 threads (8 waves), BK=64
// ---------------------------------------------------------------------------
union __align__(16) SMemQKV {
    struct { _Float16 A[64 * 64]; _Float16 B[128 * 64]; } st;  // 24 KB
    float ep[64 * 128];                                        // 32 KB
};

__global__ __launch_bounds__(256) void qkv_rope_wmma(
    const float* __restrict__ x, const float* __restrict__ w,
    const float* __restrict__ bias, const float* __restrict__ cosT,
    const float* __restrict__ sinT,
    _Float16* __restrict__ Qh, _Float16* __restrict__ Kh, _Float16* __restrict__ Vh) {
    __shared__ SMemQKV sm;

    const int tid = threadIdx.x;
    const int lane = tid & 31, wave = tid >> 5;
    const int bank = lane >> 4, ln16 = lane & 15;
    const int wm = wave >> 2, wn = wave & 3;      // 2x4 wave grid -> 32x32 per wave

    const int colTile = blockIdx.x;               // 0..47 over 6144 cols
    const int rowBase = blockIdx.y * 64;          // over B*S = 4096 rows
    const int colBase = colTile * 128;

    f8 acc[2][2] = {};

    for (int k0 = 0; k0 < D_MODEL; k0 += 64) {
        { // stage A: x rows fp32 -> fp16, 64x64
            int r = tid >> 2, c = (tid & 3) * 16;
            const float4* g =
                reinterpret_cast<const float4*>(&x[(size_t)(rowBase + r) * D_MODEL + k0 + c]);
            _Float16* dst = &sm.st.A[r * 64 + c];
#pragma unroll
            for (int j = 0; j < 4; ++j) {
                float4 v = g[j];
                dst[4 * j + 0] = (_Float16)v.x; dst[4 * j + 1] = (_Float16)v.y;
                dst[4 * j + 2] = (_Float16)v.z; dst[4 * j + 3] = (_Float16)v.w;
            }
        }
        { // stage B transposed: W[k][n] -> Bl[n*64 + k]; pack k-pairs as b32 stores
            int p = tid & 31;                 // kk = 2p, 2p+1
            int n0 = (tid >> 5) * 16;
            const float4* g0 = reinterpret_cast<const float4*>(
                &w[(size_t)(k0 + 2 * p) * THREE_D + colBase + n0]);
            const float4* g1 = reinterpret_cast<const float4*>(
                &w[(size_t)(k0 + 2 * p + 1) * THREE_D + colBase + n0]);
#pragma unroll
            for (int j = 0; j < 4; ++j) {
                float4 a = g0[j], b = g1[j];
                h2 t;
                t[0] = (_Float16)a.x; t[1] = (_Float16)b.x;
                *reinterpret_cast<h2*>(&sm.st.B[(n0 + 4 * j + 0) * 64 + 2 * p]) = t;
                t[0] = (_Float16)a.y; t[1] = (_Float16)b.y;
                *reinterpret_cast<h2*>(&sm.st.B[(n0 + 4 * j + 1) * 64 + 2 * p]) = t;
                t[0] = (_Float16)a.z; t[1] = (_Float16)b.z;
                *reinterpret_cast<h2*>(&sm.st.B[(n0 + 4 * j + 2) * 64 + 2 * p]) = t;
                t[0] = (_Float16)a.w; t[1] = (_Float16)b.w;
                *reinterpret_cast<h2*>(&sm.st.B[(n0 + 4 * j + 3) * 64 + 2 * p]) = t;
            }
        }
        __syncthreads();
#pragma unroll
        for (int kc = 0; kc < 2; ++kc) {
            h16 a0 = ldsFrag(sm.st.A, wm * 32 + ln16,      64, kc * 32, bank);
            h16 a1 = ldsFrag(sm.st.A, wm * 32 + 16 + ln16, 64, kc * 32, bank);
            h16 b0 = ldsFrag(sm.st.B, wn * 32 + ln16,      64, kc * 32, bank);
            h16 b1 = ldsFrag(sm.st.B, wn * 32 + 16 + ln16, 64, kc * 32, bank);
            acc[0][0] = wmma32(a0, b0, acc[0][0]);
            acc[0][1] = wmma32(a0, b1, acc[0][1]);
            acc[1][0] = wmma32(a1, b0, acc[1][0]);
            acc[1][1] = wmma32(a1, b1, acc[1][1]);
        }
        __syncthreads();
    }

    // spill f32 tile to LDS so RoPE can pair d <-> d^64 across waves
#pragma unroll
    for (int mi = 0; mi < 2; ++mi)
#pragma unroll
        for (int ni = 0; ni < 2; ++ni)
#pragma unroll
            for (int r = 0; r < 8; ++r)
                sm.ep[(wm * 32 + mi * 16 + r + 8 * bank) * 128 + wn * 32 + ni * 16 + ln16] =
                    acc[mi][ni][r];
    __syncthreads();

    const int kind = colTile >> 4;   // 0=Q 1=K 2=V
    const int head = colTile & 15;
    const float qscale = 0.08838834764831845f;   // 1/sqrt(128)
    for (int e = tid; e < 64 * 128; e += 256) {
        int i = e >> 7, d = e & 127;
        int grow = rowBase + i;
        int b = grow >> 11, s = grow & 2047;
        float v = sm.ep[i * 128 + d] + bias[colBase + d];
        size_t dst = (((size_t)(b * NH + head)) * SEQ + s) * HD + d;
        if (kind == 2) {
            Vh[dst] = (_Float16)v;
        } else {
            int pd = (d < 64) ? d + 64 : d - 64;
            float pv = sm.ep[i * 128 + pd] + bias[colBase + pd];
            float rot = (d < 64) ? -pv : pv;
            float rv = v * cosT[s * HD + d] + rot * sinT[s * HD + d];
            if (kind == 0) Qh[dst] = (_Float16)(rv * qscale);
            else           Kh[dst] = (_Float16)rv;
        }
    }
}

// ---------------------------------------------------------------------------
// Kernel 2: causal flash attention per (b,h,qtile). 128 threads = 4 waves,
// each wave owns 16 query rows. Q/K tiles staged via async-to-LDS copies.
// ---------------------------------------------------------------------------
__global__ __launch_bounds__(128) void flash_attn_wmma(
    const _Float16* __restrict__ Qh, const _Float16* __restrict__ Kh,
    const _Float16* __restrict__ Vh, _Float16* __restrict__ AO) {
    __shared__ __align__(16) _Float16 Ql[64 * 128];
    __shared__ __align__(16) _Float16 Kl[64 * 128];
    __shared__ __align__(16) _Float16 Vt[128 * 64];   // transposed V tile
    __shared__ __align__(16) _Float16 Pl[4][16 * 64]; // per-wave P staging

    const int tid = threadIdx.x;
    const int lane = tid & 31, wave = tid >> 5;
    const int bank = lane >> 4, ln16 = lane & 15;
    const int qi = blockIdx.x;            // 32 query tiles
    const int bh = blockIdx.y;            // b*16 + h
    const int b = bh >> 4, h = bh & 15;
    const size_t headBase = (size_t)bh * SEQ * HD;
    const int qbase = qi * 64;

    // Q tile: 64 rows x 128 halves, async global->LDS
    for (int e = tid; e < 1024; e += 128) {
        int row = e >> 4, c8 = (e & 15) * 8;
        asyncCopy16(&Ql[row * 128 + c8],
                    &Qh[headBase + (size_t)(qbase + row) * HD + c8]);
    }

    float m_i[8], l_i[8];
    f8 o[8] = {};
#pragma unroll
    for (int r = 0; r < 8; ++r) { m_i[r] = -3.0e38f; l_i[r] = 0.f; }
    asyncWait();
    __syncthreads();

    for (int j = 0; j <= qi; ++j) {
        const int kbase = j * 64;
        // K tile async; V tile staged transposed manually (copy cannot transpose)
        for (int e = tid; e < 1024; e += 128) {
            int row = e >> 4, c8 = (e & 15) * 8;
            asyncCopy16(&Kl[row * 128 + c8],
                        &Kh[headBase + (size_t)(kbase + row) * HD + c8]);
            h8 v = *reinterpret_cast<const h8*>(&Vh[headBase + (size_t)(kbase + row) * HD + c8]);
#pragma unroll
            for (int t = 0; t < 8; ++t) Vt[(c8 + t) * 64 + row] = v[t];
        }
        if (j < qi) {  // prefetch next key block while this one computes
            __builtin_prefetch(&Kh[headBase + (size_t)(kbase + 64) * HD + tid * 64], 0, 0);
            __builtin_prefetch(&Vh[headBase + (size_t)(kbase + 64) * HD + tid * 64], 0, 0);
        }
        asyncWait();
        __syncthreads();

        // S = Q K^T : 16 wmma per wave
        f8 sfr[4] = {};
#pragma unroll
        for (int kc = 0; kc < 4; ++kc) {
            h16 a = ldsFrag(Ql, wave * 16 + ln16, 128, kc * 32, bank);
#pragma unroll
            for (int f = 0; f < 4; ++f) {
                h16 bf = ldsFrag(Kl, f * 16 + ln16, 128, kc * 32, bank);
                sfr[f] = wmma32(a, bf, sfr[f]);
            }
        }
        if (j == qi) {  // causal mask on diagonal block only
#pragma unroll
            for (int f = 0; f < 4; ++f) {
                int key = kbase + f * 16 + ln16;
#pragma unroll
                for (int r = 0; r < 8; ++r)
                    if (key > qbase + wave * 16 + r + 8 * bank) sfr[f][r] = -3.0e38f;
            }
        }
        // online softmax: each row lives in a 16-lane group (same bank)
#pragma unroll
        for (int r = 0; r < 8; ++r) {
            float mx = fmaxf(fmaxf(sfr[0][r], sfr[1][r]), fmaxf(sfr[2][r], sfr[3][r]));
#pragma unroll
            for (int off = 8; off >= 1; off >>= 1) mx = fmaxf(mx, __shfl_xor(mx, off, 32));
            float mnew = fmaxf(m_i[r], mx);
            float alpha = __expf(m_i[r] - mnew);
            m_i[r] = mnew;
            float rs = 0.f;
#pragma unroll
            for (int f = 0; f < 4; ++f) {
                float p = __expf(sfr[f][r] - mnew);
                sfr[f][r] = p;
                rs += p;
            }
#pragma unroll
            for (int off = 8; off >= 1; off >>= 1) rs += __shfl_xor(rs, off, 32);
            l_i[r] = l_i[r] * alpha + rs;
#pragma unroll
            for (int f = 0; f < 8; ++f) o[f][r] = o[f][r] * alpha;
        }
        // reshape P via wave-private LDS into A-fragments, then O += P @ V
#pragma unroll
        for (int f = 0; f < 4; ++f)
#pragma unroll
            for (int r = 0; r < 8; ++r)
                Pl[wave][(r + 8 * bank) * 64 + f * 16 + ln16] = (_Float16)sfr[f][r];
#pragma unroll
        for (int kc = 0; kc < 2; ++kc) {
            h16 a = ldsFrag(Pl[wave], ln16, 64, kc * 32, bank);
#pragma unroll
            for (int f = 0; f < 8; ++f) {
                h16 bf = ldsFrag(Vt, f * 16 + ln16, 64, kc * 32, bank);
                o[f] = wmma32(a, bf, o[f]);
            }
        }
        __syncthreads();
    }

    // normalize and write [B,S,H*HD] fp16 for the out-projection
#pragma unroll
    for (int f = 0; f < 8; ++f)
#pragma unroll
        for (int r = 0; r < 8; ++r) {
            int qrow = qbase + wave * 16 + r + 8 * bank;
            float val = o[f][r] / l_i[r];
            AO[((size_t)(b * SEQ + qrow)) * D_MODEL + h * HD + f * 16 + ln16] = (_Float16)val;
        }
}

// ---------------------------------------------------------------------------
// Kernel 3: out = AO @ out_w + out_b (fp32 result). A tile async, BK=64.
// ---------------------------------------------------------------------------
__global__ __launch_bounds__(256) void outproj_wmma(
    const _Float16* __restrict__ AO, const float* __restrict__ w,
    const float* __restrict__ bias, float* __restrict__ out) {
    __shared__ __align__(16) _Float16 Al[64 * 64];    // 8 KB
    __shared__ __align__(16) _Float16 Bl[128 * 64];   // 16 KB

    const int tid = threadIdx.x;
    const int lane = tid & 31, wave = tid >> 5;
    const int bank = lane >> 4, ln16 = lane & 15;
    const int wm = wave >> 2, wn = wave & 3;
    const int rowBase = blockIdx.y * 64;
    const int colBase = blockIdx.x * 128;

    f8 acc[2][2] = {};
    for (int k0 = 0; k0 < D_MODEL; k0 += 64) {
        // A tile already fp16: async global->LDS (64 rows x 64 halves)
        for (int c = tid; c < 512; c += 256) {
            int row = c >> 3, off = (c & 7) * 8;
            asyncCopy16(&Al[row * 64 + off],
                        &AO[(size_t)(rowBase + row) * D_MODEL + k0 + off]);
        }
        { // B transposed fp32 -> fp16, packed k-pair b32 stores
            int p = tid & 31;
            int n0 = (tid >> 5) * 16;
            const float4* g0 = reinterpret_cast<const float4*>(
                &w[(size_t)(k0 + 2 * p) * D_MODEL + colBase + n0]);
            const float4* g1 = reinterpret_cast<const float4*>(
                &w[(size_t)(k0 + 2 * p + 1) * D_MODEL + colBase + n0]);
#pragma unroll
            for (int j = 0; j < 4; ++j) {
                float4 a = g0[j], b = g1[j];
                h2 t;
                t[0] = (_Float16)a.x; t[1] = (_Float16)b.x;
                *reinterpret_cast<h2*>(&Bl[(n0 + 4 * j + 0) * 64 + 2 * p]) = t;
                t[0] = (_Float16)a.y; t[1] = (_Float16)b.y;
                *reinterpret_cast<h2*>(&Bl[(n0 + 4 * j + 1) * 64 + 2 * p]) = t;
                t[0] = (_Float16)a.z; t[1] = (_Float16)b.z;
                *reinterpret_cast<h2*>(&Bl[(n0 + 4 * j + 2) * 64 + 2 * p]) = t;
                t[0] = (_Float16)a.w; t[1] = (_Float16)b.w;
                *reinterpret_cast<h2*>(&Bl[(n0 + 4 * j + 3) * 64 + 2 * p]) = t;
            }
        }
        asyncWait();
        __syncthreads();
#pragma unroll
        for (int kc = 0; kc < 2; ++kc) {
            h16 a0 = ldsFrag(Al, wm * 32 + ln16,      64, kc * 32, bank);
            h16 a1 = ldsFrag(Al, wm * 32 + 16 + ln16, 64, kc * 32, bank);
            h16 b0 = ldsFrag(Bl, wn * 32 + ln16,      64, kc * 32, bank);
            h16 b1 = ldsFrag(Bl, wn * 32 + 16 + ln16, 64, kc * 32, bank);
            acc[0][0] = wmma32(a0, b0, acc[0][0]);
            acc[0][1] = wmma32(a0, b1, acc[0][1]);
            acc[1][0] = wmma32(a1, b0, acc[1][0]);
            acc[1][1] = wmma32(a1, b1, acc[1][1]);
        }
        __syncthreads();
    }
#pragma unroll
    for (int mi = 0; mi < 2; ++mi)
#pragma unroll
        for (int ni = 0; ni < 2; ++ni)
#pragma unroll
            for (int r = 0; r < 8; ++r) {
                int row = rowBase + wm * 32 + mi * 16 + r + 8 * bank;
                int col = colBase + wn * 32 + ni * 16 + ln16;
                out[(size_t)row * D_MODEL + col] = acc[mi][ni][r] + bias[col];
            }
}

extern "C" void kernel_launch(void* const* d_in, const int* in_sizes, int n_in,
                              void* d_out, int out_size, void* d_ws, size_t ws_size,
                              hipStream_t stream) {
    (void)in_sizes; (void)n_in; (void)out_size; (void)ws_size;
    const float* x     = (const float*)d_in[0];
    const float* cosT  = (const float*)d_in[1];
    const float* sinT  = (const float*)d_in[2];
    const float* qkv_w = (const float*)d_in[3];
    const float* qkv_b = (const float*)d_in[4];
    const float* out_w = (const float*)d_in[5];
    const float* out_b = (const float*)d_in[6];
    float* out = (float*)d_out;

    const size_t bufElems = (size_t)2 * NH * SEQ * HD;  // 8 Mi halves = 16 MiB
    _Float16* Qh = (_Float16*)d_ws;
    _Float16* Kh = Qh + bufElems;
    _Float16* Vh = Kh + bufElems;
    _Float16* AO = Vh + bufElems;   // total 64 MiB of workspace

    qkv_rope_wmma<<<dim3(48, 64), 256, 0, stream>>>(x, qkv_w, qkv_b, cosT, sinT,
                                                    Qh, Kh, Vh);
    flash_attn_wmma<<<dim3(32, 32), 128, 0, stream>>>(Qh, Kh, Vh, AO);
    outproj_wmma<<<dim3(16, 64), 256, 0, stream>>>(AO, out_w, out_b, out);
}
